// CombinedLoss_9938554323333
// MI455X (gfx1250) — matrix operand order repaired and checked by reading, not compiled, not run on hardware
//
#include <hip/hip_runtime.h>
#include <hip/hip_bf16.h>
#include <cfloat>

// ---------------------------------------------------------------------------
// CombinedLoss (CE + batch-hard triplet) for MI455X / gfx1250 (wave32, WMMA)
//   B = 8192, D = 256, C = 1000
// Gram/distance kernel: async global->LDS staging (ASYNCcnt) + LDS-fed WMMA.
// ---------------------------------------------------------------------------

#define BB 8192
#define DD 256
#define CC 1000
#define MARGIN 0.3f

typedef __attribute__((ext_vector_type(16))) __bf16 v16bf;
typedef __attribute__((ext_vector_type(8)))  float  v8f;

union Frag16 {
    v16bf        v;
    unsigned int u[8];
};

__device__ __forceinline__ unsigned short f32_to_bf16_rne(float f) {
    unsigned int u = __float_as_uint(f);
    u += 0x7FFFu + ((u >> 16) & 1u);        // round-to-nearest-even
    return (unsigned short)(u >> 16);
}
__device__ __forceinline__ float bf16_to_f32(unsigned short h) {
    return __uint_as_float(((unsigned int)h) << 16);
}

// Two 16-byte loads -> one 16x bf16 WMMA fragment. Works for global or LDS
// (generic) pointers; offsets are element counts, 16B-aligned by construction.
__device__ __forceinline__ v16bf load_frag16(const unsigned short* p,
                                             int off0, int off1) {
    Frag16 f;
    const uint4 a = *reinterpret_cast<const uint4*>(p + off0);
    const uint4 b = *reinterpret_cast<const uint4*>(p + off1);
    f.u[0] = a.x; f.u[1] = a.y; f.u[2] = a.z; f.u[3] = a.w;
    f.u[4] = b.x; f.u[5] = b.y; f.u[6] = b.z; f.u[7] = b.w;
    return f.v;
}

// Async global->LDS 16B copy (VGLOBAL GVS form): vdst = LDS byte address,
// vaddr = 32-bit byte offset, saddr = 64-bit base. Tracked by ASYNCcnt.
#define ASYNC_CP_B128(ldsaddr, goff, gbase)                                   \
    asm volatile("global_load_async_to_lds_b128 %0, %1, %2"                   \
                 :: "v"(ldsaddr), "v"(goff), "s"(gbase) : "memory")

#define WAIT_ASYNC0() asm volatile("s_wait_asynccnt 0x0" ::: "memory")

// Low 32 bits of a generic LDS pointer == LDS byte offset (hardware truncates
// flat LDS addresses to addr[31:0]).
__device__ __forceinline__ unsigned lds_addr32(const void* p) {
    return (unsigned)(unsigned long long)p;
}

// ---------------------------------------------------------------------------
// Kernel 1: per-row prep — split-bf16 conversion, exact squared norms,
//           init hardest-pos (max, 0.0f bits) / hardest-neg (min, +inf bits).
// ---------------------------------------------------------------------------
__global__ void __launch_bounds__(DD)
prep_kernel(const float* __restrict__ feat,
            unsigned short* __restrict__ Xhi,
            unsigned short* __restrict__ Xlo,
            float* __restrict__ sqnorm,
            unsigned int* __restrict__ hp_bits,
            unsigned int* __restrict__ hn_bits) {
    __shared__ float red[DD];
    const int r = blockIdx.x;
    const int t = threadIdx.x;
    const float x = feat[r * DD + t];
    const unsigned short hb = f32_to_bf16_rne(x);
    const float xh = bf16_to_f32(hb);
    const unsigned short lb = f32_to_bf16_rne(x - xh);
    Xhi[r * DD + t] = hb;
    Xlo[r * DD + t] = lb;
    red[t] = x * x;
    __syncthreads();
    for (int s = DD / 2; s > 0; s >>= 1) {
        if (t < s) red[t] += red[t + s];
        __syncthreads();
    }
    if (t == 0) {
        sqnorm[r]  = red[0];
        hp_bits[r] = 0u;           // dist >= 0 and diagonal is always pos
        hn_bits[r] = 0x7F800000u;  // +inf  -> "no negative seen"
    }
}

// ---------------------------------------------------------------------------
// Kernel 2: cross-entropy per row (numerically stable logsumexp).
// ---------------------------------------------------------------------------
__global__ void __launch_bounds__(256)
ce_kernel(const float* __restrict__ outputs,
          const int* __restrict__ targets,
          float* __restrict__ ce_rows) {
    __shared__ float red[256];
    const int r = blockIdx.x;
    const int t = threadIdx.x;
    const float* row = outputs + (size_t)r * CC;

    float mx = -FLT_MAX;
    for (int c = t; c < CC; c += 256) mx = fmaxf(mx, row[c]);
    red[t] = mx;
    __syncthreads();
    for (int s = 128; s > 0; s >>= 1) {
        if (t < s) red[t] = fmaxf(red[t], red[t + s]);
        __syncthreads();
    }
    mx = red[0];
    __syncthreads();

    float sum = 0.f;
    for (int c = t; c < CC; c += 256) sum += expf(row[c] - mx);
    red[t] = sum;
    __syncthreads();
    for (int s = 128; s > 0; s >>= 1) {
        if (t < s) red[t] += red[t + s];
        __syncthreads();
    }
    if (t == 0) {
        const int tg = targets[r];
        ce_rows[r] = -(row[tg] - mx - logf(red[0]));
    }
}

// ---------------------------------------------------------------------------
// Kernel 3: fused Gram/distance/batch-hard mining.
// Block = 8 waves covering a 64x128 region (4 row-tiles x 2 col-supertiles).
// Per k-step: async-copy A(64x32) and B(128x32) hi/lo tiles to LDS (24 KB),
// then each wave runs 12 split-bf16 WMMAs fed by ds_load_b128.
// grid = 8192 blocks of 256 threads.
// ---------------------------------------------------------------------------
__global__ void __launch_bounds__(256)
gram_kernel(const unsigned short* __restrict__ Xhi,
            const unsigned short* __restrict__ Xlo,
            const float* __restrict__ sqnorm,
            const int* __restrict__ targets,
            unsigned int* __restrict__ hp_bits,
            unsigned int* __restrict__ hn_bits) {
    __shared__ unsigned short sAhi[64 * 32];
    __shared__ unsigned short sAlo[64 * 32];
    __shared__ unsigned short sBhi[128 * 32];
    __shared__ unsigned short sBlo[128 * 32];

    const int tid  = threadIdx.x;
    const int bid  = blockIdx.x;
    const int rb   = bid >> 6;          // 128 row-blocks of 64 rows
    const int cb   = bid & 63;          // 64 col-blocks of 128 cols
    const int i0b  = rb * 64;
    const int j0b  = cb * 128;

    const int wv   = tid >> 5;
    const int lane = tid & 31;
    const int lm   = lane & 15;         // lane-in-group
    const int h    = lane >> 4;         // lane half (0/1)
    const int wr   = wv >> 1;           // wave row-tile 0..3
    const int wc   = wv & 1;            // wave col-supertile 0..1
    const int i0   = i0b + wr * 16;
    const int j0   = j0b + wc * 64;

    const unsigned long long ghi = (unsigned long long)Xhi;
    const unsigned long long glo = (unsigned long long)Xlo;

    // Copy-phase thread mapping: A = 64 rows x 4 chunks of 8; B = 128 x 2x2.
    const int ar  = tid >> 2, ac  = tid & 3;
    const int br  = tid >> 1, bc0 = (tid & 1) * 2;
    const unsigned ldsAhi  = lds_addr32(&sAhi[ar * 32 + ac * 8]);
    const unsigned ldsAlo  = lds_addr32(&sAlo[ar * 32 + ac * 8]);
    const unsigned ldsBhi0 = lds_addr32(&sBhi[br * 32 + bc0 * 8]);
    const unsigned ldsBlo0 = lds_addr32(&sBlo[br * 32 + bc0 * 8]);
    const unsigned ldsBhi1 = ldsBhi0 + 16;
    const unsigned ldsBlo1 = ldsBlo0 + 16;

    const v8f z = {0.f, 0.f, 0.f, 0.f, 0.f, 0.f, 0.f, 0.f};
    v8f acc[4];
#pragma unroll
    for (int ct = 0; ct < 4; ++ct) acc[ct] = z;

    for (int k0 = 0; k0 < DD; k0 += 32) {
        // ---- stage tiles: 6 async b128 copies per thread ----
        const unsigned ga  = (unsigned)(((i0b + ar) * DD + k0 + ac * 8) * 2);
        const unsigned gb0 = (unsigned)(((j0b + br) * DD + k0 + bc0 * 8) * 2);
        ASYNC_CP_B128(ldsAhi,  ga,       ghi);
        ASYNC_CP_B128(ldsAlo,  ga,       glo);
        ASYNC_CP_B128(ldsBhi0, gb0,      ghi);
        ASYNC_CP_B128(ldsBlo0, gb0,      glo);
        ASYNC_CP_B128(ldsBhi1, gb0 + 16, ghi);
        ASYNC_CP_B128(ldsBlo1, gb0 + 16, glo);
        WAIT_ASYNC0();
        __syncthreads();

        // ---- consume: fragments from LDS, 12 WMMAs ----
        // A fragment (16x32): lane half h holds K {h*8..h*8+7, h*8+16..h*8+23}
        const int arow = wr * 16 + lm;
        const int aoff = arow * 32 + h * 8;
        const v16bf fahi = load_frag16(sAhi, aoff, aoff + 16);
        const v16bf falo = load_frag16(sAlo, aoff, aoff + 16);
#pragma unroll
        for (int ct = 0; ct < 4; ++ct) {
            // B fragment (32x16): column = staged row (wc*64+ct*16+lm),
            // K {h*8..h*8+7, h*8+16..h*8+23} (matches A).
            const int brow = wc * 64 + ct * 16 + lm;
            const int boff = brow * 32 + h * 8;
            const v16bf fbhi = load_frag16(sBhi, boff, boff + 16);
            const v16bf fblo = load_frag16(sBlo, boff, boff + 16);
            // 3xBF16 split product: hi*hi + hi*lo + lo*hi (~f32 accuracy)
            acc[ct] = __builtin_amdgcn_wmma_f32_16x16x32_bf16(
                false, fahi, false, fbhi, (short)0, acc[ct], false, false);
            acc[ct] = __builtin_amdgcn_wmma_f32_16x16x32_bf16(
                false, fahi, false, fblo, (short)0, acc[ct], false, false);
            acc[ct] = __builtin_amdgcn_wmma_f32_16x16x32_bf16(
                false, falo, false, fbhi, (short)0, acc[ct], false, false);
        }
        __syncthreads();   // all waves done with LDS before next overwrite
    }

    // ---- epilogue: distances + masked max/min; combine the 4 column tiles
    // in-register, 16-lane shuffle tree, then 1 atomic pair per row. ----
    float sqj[4];
    int   tj[4];
#pragma unroll
    for (int ct = 0; ct < 4; ++ct) {
        const int gj = j0 + ct * 16 + lm;
        sqj[ct] = sqnorm[gj];
        tj[ct]  = targets[gj];
    }

#pragma unroll
    for (int p = 0; p < 8; ++p) {
        // C/D layout: VGPR p, lanes 0-15 -> M=p, lanes 16-31 -> M=p+8; N=lm
        const int gi = i0 + p + h * 8;
        const float sqi = sqnorm[gi];
        const int   ti  = targets[gi];
        float pv = 0.f;        // hardest positive (max); >=0 always holds
        float nv = FLT_MAX;    // hardest negative (min)
#pragma unroll
        for (int ct = 0; ct < 4; ++ct) {
            const float d2 = sqi + sqj[ct] - 2.f * acc[ct][p];
            const float dist = (d2 > 0.f) ? sqrtf(d2) : 0.f;
            if (ti == tj[ct]) pv = fmaxf(pv, dist);
            else              nv = fminf(nv, dist);
        }
#pragma unroll
        for (int m = 1; m < 16; m <<= 1) {
            pv = fmaxf(pv, __shfl_xor(pv, m, 16));
            nv = fminf(nv, __shfl_xor(nv, m, 16));
        }
        if (lm == 0) {
            // non-negative floats: IEEE bit pattern order == unsigned order
            atomicMax(&hp_bits[gi], __float_as_uint(pv));
            atomicMin(&hn_bits[gi], __float_as_uint(nv));
        }
    }
}

// ---------------------------------------------------------------------------
// Kernel 4: deterministic final reduction -> (total, ce, triplet).
// ---------------------------------------------------------------------------
__global__ void __launch_bounds__(256)
finalize_kernel(const float* __restrict__ ce_rows,
                const unsigned int* __restrict__ hp_bits,
                const unsigned int* __restrict__ hn_bits,
                float* __restrict__ out) {
    __shared__ float rc[256];
    __shared__ float rt[256];
    const int t = threadIdx.x;
    float ces = 0.f, ts = 0.f;
    for (int r = t; r < BB; r += 256) {
        ces += ce_rows[r];
        const unsigned int hnb = hn_bits[r];
        if (hnb != 0x7F800000u) {   // a negative exists -> row valid
            const float hpv = __uint_as_float(hp_bits[r]);
            const float hnv = __uint_as_float(hnb);
            ts += fmaxf(hpv - hnv + MARGIN, 0.f);
        }
    }
    rc[t] = ces;
    rt[t] = ts;
    __syncthreads();
    for (int s = 128; s > 0; s >>= 1) {
        if (t < s) { rc[t] += rc[t + s]; rt[t] += rt[t + s]; }
        __syncthreads();
    }
    if (t == 0) {
        const float ce = rc[0] / (float)BB;
        const float tr = rt[0] / (float)BB;
        out[0] = ce + tr;   // total
        out[1] = ce;        // ce_loss
        out[2] = tr;        // triplet_loss
    }
}

// ---------------------------------------------------------------------------
// Host-side launch
// ---------------------------------------------------------------------------
extern "C" void kernel_launch(void* const* d_in, const int* in_sizes, int n_in,
                              void* d_out, int out_size, void* d_ws, size_t ws_size,
                              hipStream_t stream) {
    const float* outputs  = (const float*)d_in[0];   // [8192, 1000]
    const float* features = (const float*)d_in[1];   // [8192, 256]
    const int*   targets  = (const int*)d_in[2];     // [8192]
    float*       out      = (float*)d_out;           // [3]

    char* ws = (char*)d_ws;
    unsigned short* Xhi     = (unsigned short*)(ws);                     // 4 MB
    unsigned short* Xlo     = (unsigned short*)(ws + 4194304);           // 4 MB
    float*          sqnorm  = (float*)        (ws + 8388608);            // 32 KB
    unsigned int*   hp_bits = (unsigned int*) (ws + 8388608 + 32768);    // 32 KB
    unsigned int*   hn_bits = (unsigned int*) (ws + 8388608 + 65536);    // 32 KB
    float*          ce_rows = (float*)        (ws + 8388608 + 98304);    // 32 KB

    prep_kernel<<<BB, DD, 0, stream>>>(features, Xhi, Xlo, sqnorm, hp_bits, hn_bits);
    ce_kernel<<<BB, 256, 0, stream>>>(outputs, targets, ce_rows);
    gram_kernel<<<BB, 256, 0, stream>>>(Xhi, Xlo, sqnorm, targets, hp_bits, hn_bits);
    finalize_kernel<<<1, 256, 0, stream>>>(ce_rows, hp_bits, hn_bits, out);
}
